// MoELayer_23244363005985
// MI455X (gfx1250) — matrix operand order repaired and checked by reading, not compiled, not run on hardware
//
#include <hip/hip_runtime.h>
#include <math.h>

typedef __attribute__((ext_vector_type(16))) _Float16     v16h;
typedef __attribute__((ext_vector_type(8)))  float        v8f;
typedef __attribute__((ext_vector_type(4)))  float        f32x4;
typedef __attribute__((ext_vector_type(4)))  unsigned int u32x4;

#define D_MODEL 1024
#define HIDDEN  4096
#define NEXP    8
#define TOPK    2
#define DEPTH_SCALE 0.28867513459481287f  /* 1/sqrt(12) */

#define MT     16    /* token tile (WMMA M) */
#define HC     64    /* H chunk per iteration */
#define NBLK   16    /* 16 N-blocks of 16 -> 256 output cols per wave */
#define WCOLS  256
#define XPAD   8     /* (1024+8)*2 = 2064 B row stride, 16B multiple */
#define HPAD   8     /* (64+8)*2   = 144 B row stride, 16B multiple  */
#define WROW   20    /* packed-dword row stride (16 + pad), 80 B = 16B multiple */

union Frag {
  v16h v;
  unsigned int u[8];
  u32x4 q[2];
};
union PackPair { _Float16 h[2]; unsigned int u; };

// ---------------------------------------------------------------------------
// Kernel 1: router -- logits, top-2 softmax, counts, and fp32 -> f16 convert
// ---------------------------------------------------------------------------
__global__ __launch_bounds__(128) void router_kernel(
    const float* __restrict__ x, const float* __restrict__ Wr,
    _Float16* __restrict__ xh, int* __restrict__ top_e,
    float* __restrict__ top_w, int* __restrict__ counts, int ntok)
{
  const int wave = threadIdx.x >> 5;
  const int lane = threadIdx.x & 31;
  const int tok = blockIdx.x * 4 + wave;
  if (tok >= ntok) return;

  const float* xr = x + (size_t)tok * D_MODEL;
  _Float16* xo = xh + (size_t)tok * D_MODEL;

  float l[NEXP];
#pragma unroll
  for (int e = 0; e < NEXP; ++e) l[e] = 0.f;

  for (int c = lane; c < D_MODEL; c += 32) {
    float xv = xr[c];
    xo[c] = (_Float16)xv;
#pragma unroll
    for (int e = 0; e < NEXP; ++e) l[e] += xv * Wr[c * NEXP + e];
  }
#pragma unroll
  for (int e = 0; e < NEXP; ++e) {
#pragma unroll
    for (int off = 16; off > 0; off >>= 1)
      l[e] += __shfl_xor(l[e], off, 32);
  }

  if (lane == 0) {
    int e1 = 0; float v1 = l[0];
#pragma unroll
    for (int e = 1; e < NEXP; ++e) if (l[e] > v1) { v1 = l[e]; e1 = e; }
    int e2 = (e1 == 0) ? 1 : 0; float v2 = l[e2];
#pragma unroll
    for (int e = 0; e < NEXP; ++e)
      if (e != e1 && l[e] > v2) { v2 = l[e]; e2 = e; }
    float w1 = 1.f / (1.f + __expf(v2 - v1));   /* softmax over {v1,v2}, v1 >= v2 */
    top_e[2 * tok]     = e1;
    top_e[2 * tok + 1] = e2;
    top_w[2 * tok]     = w1;
    top_w[2 * tok + 1] = 1.f - w1;
    atomicAdd(&counts[e1], 1);
    atomicAdd(&counts[e2], 1);
  }
}

// ---------------------------------------------------------------------------
// Kernel 2: tiny exclusive scan over 8 expert counts
// ---------------------------------------------------------------------------
__global__ void scan_kernel(const int* __restrict__ counts,
                            int* __restrict__ offsets,
                            int* __restrict__ cursor)
{
  if (threadIdx.x == 0) {
    int s = 0;
    for (int e = 0; e < NEXP; ++e) {
      offsets[e] = s;
      cursor[e]  = s;
      s += counts[e];
    }
    offsets[NEXP] = s;
  }
}

// ---------------------------------------------------------------------------
// Kernel 3: scatter tokens into expert-sorted segments
// ---------------------------------------------------------------------------
__global__ __launch_bounds__(128) void scatter_kernel(
    const int* __restrict__ top_e, const float* __restrict__ top_w,
    int* __restrict__ cursor, int* __restrict__ tok_idx,
    float* __restrict__ tok_w, int ntok)
{
  int t = blockIdx.x * 128 + threadIdx.x;
  if (t >= ntok) return;
#pragma unroll
  for (int k = 0; k < TOPK; ++k) {
    int e = top_e[2 * t + k];
    int pos = atomicAdd(&cursor[e], 1);
    tok_idx[pos] = t;
    tok_w[pos]   = top_w[2 * t + k];
  }
}

// ---------------------------------------------------------------------------
// Kernel 4: gathered SwiGLU expert MLP via f16 WMMA, fp32 accumulate
//   grid.x = NEXP * 256 ; block = 128 threads (4 waves)
//   workgroup: 16 tokens of one expert, full 1024 output columns
//   B operands staged through wave-private LDS transpose buffers so that
//   fragments load as 2x ds_load_b128 and global traffic is b128-coalesced.
// ---------------------------------------------------------------------------
__global__ __launch_bounds__(128) void moe_gemm_kernel(
    const float* __restrict__ Wg, const float* __restrict__ Wu,
    const float* __restrict__ Wd, const _Float16* __restrict__ xh,
    const int* __restrict__ offsets, const int* __restrict__ tok_idx,
    const float* __restrict__ tok_w, float* __restrict__ out)
{
  __shared__ _Float16 xs[MT][D_MODEL + XPAD];       /* x tile, f16  (~33 KB) */
  __shared__ _Float16 hs[MT][HC + HPAD];            /* h chunk, f16          */
  __shared__ unsigned int wbuf[4][2][MT][WROW];     /* wave-private B stage  */
  __shared__ int   s_tok[MT];
  __shared__ float s_w[MT];

  const int e    = blockIdx.x >> 8;     /* 256 tiles per expert */
  const int tile = blockIdx.x & 255;
  const int seg_beg = offsets[e];
  const int seg_end = offsets[e + 1];
  const int m0 = seg_beg + tile * MT;
  if (m0 >= seg_end) return;            /* uniform per block */
  const int rows = (seg_end - m0 < MT) ? (seg_end - m0) : MT;

  const int tid  = threadIdx.x;
  const int wave = tid >> 5;
  const int lane = tid & 31;
  const int half = lane >> 4;
  const int idx  = lane & 15;
  const int kp0  = lane & 15;           /* staging: k-pair index   */
  const int cg0  = lane >> 4;           /* staging: col-group base */

  if (tid < MT) {
    int tok = (tid < rows) ? tok_idx[m0 + tid] : -1;
    s_tok[tid] = tok;
    s_w[tid]   = (tid < rows) ? tok_w[m0 + tid] : 0.f;
  }
  __syncthreads();

  /* stage gathered x rows (f16) into LDS, dword-wide, zero-pad missing rows */
  for (int i = tid; i < MT * (D_MODEL / 2); i += 128) {
    int r  = i >> 9;           /* /512 dword-pairs per row */
    int cp = i & 511;
    int tok = s_tok[r];
    unsigned int v = 0u;
    if (tok >= 0)
      v = ((const unsigned int*)(xh + (size_t)tok * D_MODEL))[cp];
    *(unsigned int*)&xs[r][cp * 2] = v;
  }
  __syncthreads();

  const float* Wge = Wg + (size_t)e * D_MODEL * HIDDEN;
  const float* Wue = Wu + (size_t)e * D_MODEL * HIDDEN;
  const float* Wde = Wd + (size_t)e * HIDDEN * D_MODEL;

  v8f accd[NBLK];
#pragma unroll
  for (int b = 0; b < NBLK; ++b) {
#pragma unroll
    for (int r = 0; r < 8; ++r) accd[b][r] = 0.f;
  }

  const int colbase = wave * WCOLS;

#pragma unroll 1
  for (int ch = 0; ch < HIDDEN / HC; ++ch) {
    /* ---------------- Phase A: h = silu(x@Wg) * (x@Wu), 16x64 chunk -------- */
    const int hcol0 = ch * HC + wave * MT;   /* this wave's 16 H columns */
    if (ch + 1 < HIDDEN / HC) {              /* hint next chunk into caches */
      __builtin_prefetch(Wge + (size_t)hcol0 + HC, 0, 1);
      __builtin_prefetch(Wue + (size_t)hcol0 + HC, 0, 1);
    }
    v8f accg, accu;
#pragma unroll
    for (int r = 0; r < 8; ++r) { accg[r] = 0.f; accu[r] = 0.f; }

#pragma unroll 1
    for (int t = 0; t < D_MODEL / 32; ++t) {
      const int kt = t * 32;
      /* stage this wave's 32x16 slices of Wg/Wu, packed (k,k+1) per dword */
      const float* gbase = Wge + (size_t)kt * HIDDEN + hcol0;
      const float* ubase = Wue + (size_t)kt * HIDDEN + hcol0;
#pragma unroll
      for (int j = 0; j < 2; ++j) {
        const int kp = kp0;
        const int cg = cg0 + 2 * j;
        const float* g0 = gbase + (size_t)(2 * kp) * HIDDEN + cg * 4;
        const float* u0 = ubase + (size_t)(2 * kp) * HIDDEN + cg * 4;
        f32x4 ga = *(const f32x4*)g0;
        f32x4 gb = *(const f32x4*)(g0 + HIDDEN);
        f32x4 ua = *(const f32x4*)u0;
        f32x4 ub = *(const f32x4*)(u0 + HIDDEN);
#pragma unroll
        for (int c = 0; c < 4; ++c) {
          PackPair pg; pg.h[0] = (_Float16)ga[c]; pg.h[1] = (_Float16)gb[c];
          wbuf[wave][0][cg * 4 + c][kp] = pg.u;
          PackPair pu; pu.h[0] = (_Float16)ua[c]; pu.h[1] = (_Float16)ub[c];
          wbuf[wave][1][cg * 4 + c][kp] = pu.u;
        }
      }
      /* fragments: 2x b128 LDS reads each (A and B both in contiguous k = 16*half + s order) */
      Frag a, bg, bu;
      a.q[0]  = *(const u32x4*)&xs[idx][kt + 16 * half];
      a.q[1]  = *(const u32x4*)&xs[idx][kt + 16 * half + 8];
      bg.q[0] = *(const u32x4*)&wbuf[wave][0][idx][8 * half];
      bg.q[1] = *(const u32x4*)&wbuf[wave][0][idx][8 * half + 4];
      bu.q[0] = *(const u32x4*)&wbuf[wave][1][idx][8 * half];
      bu.q[1] = *(const u32x4*)&wbuf[wave][1][idx][8 * half + 4];
      accg = __builtin_amdgcn_wmma_f32_16x16x32_f16(
          false, a.v, false, bg.v, (short)0, accg, false, false);
      accu = __builtin_amdgcn_wmma_f32_16x16x32_f16(
          false, a.v, false, bu.v, (short)0, accu, false, false);
    }

    __syncthreads();   /* previous Phase B finished reading hs */
#pragma unroll
    for (int r = 0; r < 8; ++r) {
      float g = accg[r], u = accu[r];
      float sig = __builtin_amdgcn_rcpf(1.f + __expf(-g));  /* fast sigmoid */
      hs[8 * half + r][wave * MT + idx] = (_Float16)(g * sig * u);
    }
    __syncthreads();

    /* ---------------- Phase B: out += h_chunk @ Wd_chunk ------------------- */
#pragma unroll 1
    for (int t = 0; t < HC / 32; ++t) {
      const int kt = t * 32;
      Frag a;
      a.q[0] = *(const u32x4*)&hs[idx][kt + 16 * half];
      a.q[1] = *(const u32x4*)&hs[idx][kt + 16 * half + 8];
#pragma unroll 1
      for (int b = 0; b < NBLK; ++b) {
        const float* dbase =
            Wde + (size_t)(ch * HC + kt) * D_MODEL + colbase + 16 * b;
#pragma unroll
        for (int j = 0; j < 2; ++j) {
          const int kp = kp0;
          const int cg = cg0 + 2 * j;
          const float* d0 = dbase + (size_t)(2 * kp) * D_MODEL + cg * 4;
          f32x4 da = *(const f32x4*)d0;
          f32x4 db = *(const f32x4*)(d0 + D_MODEL);
#pragma unroll
          for (int c = 0; c < 4; ++c) {
            PackPair pd; pd.h[0] = (_Float16)da[c]; pd.h[1] = (_Float16)db[c];
            wbuf[wave][0][cg * 4 + c][kp] = pd.u;
          }
        }
        Frag bd;
        bd.q[0] = *(const u32x4*)&wbuf[wave][0][idx][8 * half];
        bd.q[1] = *(const u32x4*)&wbuf[wave][0][idx][8 * half + 4];
        accd[b] = __builtin_amdgcn_wmma_f32_16x16x32_f16(
            false, a.v, false, bd.v, (short)0, accd[b], false, false);
      }
    }
  }

  /* ---------------- epilogue: weighted scatter-accumulate ------------------ */
#pragma unroll
  for (int r = 0; r < 8; ++r) {
    const int m = 8 * half + r;
    const int tok = s_tok[m];
    if (tok >= 0) {
      const float w = s_w[m] * DEPTH_SCALE;
#pragma unroll 1
      for (int b = 0; b < NBLK; ++b) {
        const int col = colbase + 16 * b + idx;
        atomicAdd(&out[(size_t)tok * D_MODEL + col], w * accd[b][r]);
      }
    }
  }
}

// ---------------------------------------------------------------------------
extern "C" void kernel_launch(void* const* d_in, const int* in_sizes, int n_in,
                              void* d_out, int out_size, void* d_ws, size_t ws_size,
                              hipStream_t stream)
{
  (void)n_in; (void)ws_size;
  const float* x  = (const float*)d_in[0];
  const float* Wr = (const float*)d_in[1];
  const float* Wg = (const float*)d_in[2];
  const float* Wu = (const float*)d_in[3];
  const float* Wd = (const float*)d_in[4];
  float* out = (float*)d_out;

  const int ntok = in_sizes[0] / D_MODEL;   /* 4096 */

  char* p = (char*)d_ws;
  int*      counts  = (int*)p;       p += 256;
  int*      offsets = (int*)p;       p += 256;
  int*      cursor  = (int*)p;       p += 256;
  int*      top_e   = (int*)p;       p += (size_t)ntok * 2 * sizeof(int);
  float*    top_w   = (float*)p;     p += (size_t)ntok * 2 * sizeof(float);
  int*      tok_idx = (int*)p;       p += (size_t)ntok * 2 * sizeof(int);
  float*    tok_w   = (float*)p;     p += (size_t)ntok * 2 * sizeof(float);
  _Float16* xh      = (_Float16*)p;  p += (size_t)ntok * D_MODEL * sizeof(_Float16);

  hipMemsetAsync(counts, 0, 256, stream);
  hipMemsetAsync(d_out, 0, (size_t)out_size * sizeof(float), stream);

  router_kernel<<<ntok / 4, 128, 0, stream>>>(x, Wr, xh, top_e, top_w, counts, ntok);
  scan_kernel<<<1, 32, 0, stream>>>(counts, offsets, cursor);
  scatter_kernel<<<(ntok + 127) / 128, 128, 0, stream>>>(top_e, top_w, cursor,
                                                         tok_idx, tok_w, ntok);
  moe_gemm_kernel<<<NEXP * 256, 128, 0, stream>>>(Wg, Wu, Wd, xh, offsets,
                                                  tok_idx, tok_w, out);
}
